// CosmoNetHybrid_63556926046863
// MI455X (gfx1250) — matrix operand order repaired and checked by reading, not compiled, not run on hardware
//
#include <hip/hip_runtime.h>
#include <hip/hip_bf16.h>
#include <math.h>

typedef __attribute__((ext_vector_type(16))) _Float16 v16h;
typedef __attribute__((ext_vector_type(8)))  float    v8f;

// ---------------------------------------------------------------------------
// Tensor Data Mover (TDM) 2-D tile load: global -> LDS (ISA ch. 8 descriptor)
// ---------------------------------------------------------------------------
#if __has_builtin(__builtin_amdgcn_tensor_load_to_lds) && \
    __has_builtin(__builtin_amdgcn_s_wait_tensorcnt)
#define CDNA5_TDM 1
typedef unsigned int v4u  __attribute__((ext_vector_type(4)));
typedef int          v4it __attribute__((ext_vector_type(4)));
typedef int          v8it __attribute__((ext_vector_type(8)));

// dszcode: 0=1B 1=2B 2=4B 3=8B. tile0 x tile1 tile of a tensor whose dim0
// length is dim0 elements, dim1 length dim1 rows, row stride stride0 elements.
__device__ inline void tdm_load_2d(const void* src, void* lds_dst,
                                   unsigned dszcode, unsigned tile0, unsigned tile1,
                                   unsigned dim0, unsigned dim1, unsigned stride0) {
    unsigned long long ga = (unsigned long long)(size_t)src;
    unsigned loff = (unsigned)(size_t)lds_dst;       // LDS aperture: low 32 bits = LDS byte offset
    v4u g0;
    g0.x = 1u;                                       // count=1, is_restore=0
    g0.y = loff;                                     // lds_addr [63:32]
    g0.z = (unsigned)(ga & 0xffffffffull);           // global_addr [95:64]
    g0.w = (unsigned)((ga >> 32) & 0x01ffffffull)    // global_addr [120:96]
         | (2u << 30);                               // type=2 ("image") [127:126]
    v8it g1;
    g1[0] = (int)(dszcode << 16);                    // workgroup_mask=0, data_size
    g1[1] = (int)(dim0 << 16);                       // tensor_dim0 [79:48] lo
    g1[2] = (int)((dim0 >> 16) | (dim1 << 16));      // tensor_dim0 hi | tensor_dim1 lo
    g1[3] = (int)((dim1 >> 16) | (tile0 << 16));     // tensor_dim1 hi | tile_dim0 [127:112]
    g1[4] = (int)(tile1 & 0xffffu);                  // tile_dim1, tile_dim2=0
    g1[5] = (int)stride0;                            // tensor_dim0_stride [207:160] lo32
    g1[6] = 0;
    g1[7] = 0;
    v4it z4 = {0, 0, 0, 0};
#if defined(__clang_major__) && (__clang_major__ >= 23)
    v8it z8 = {0, 0, 0, 0, 0, 0, 0, 0};
    __builtin_amdgcn_tensor_load_to_lds(g0, g1, z4, z4, z8, 0);
#else
    __builtin_amdgcn_tensor_load_to_lds(g0, g1, z4, z4, 0);
#endif
}
#endif

// ---------------------------------------------------------------------------
// FFT helpers
// ---------------------------------------------------------------------------
__device__ inline float2 c_mul(float2 a, float2 b) {
    float2 r; r.x = a.x * b.x - a.y * b.y; r.y = a.x * b.y + a.y * b.x; return r;
}

// 256-pt radix-2 DIT FFT over rows. grid = B*256 blocks, 128 threads.
__global__ void fft_rows_kernel(const float* __restrict__ img, float2* __restrict__ out) {
    int b = blockIdx.x >> 8;
    int y = blockIdx.x & 255;
    int t = threadIdx.x;
    __shared__ float  s_raw[256];
    __shared__ float2 s[256];
    const float* row = img + ((size_t)b * 256 + y) * 256;
#ifdef CDNA5_TDM
    if (threadIdx.x == 0)                       // TDM ignores EXEC; issue once
        tdm_load_2d(row, s_raw, 2u, 256u, 1u, 256u, 1u, 256u);
    __builtin_amdgcn_s_wait_tensorcnt(0);
    __syncthreads();
#else
    for (int e = t; e < 256; e += 128) s_raw[e] = row[e];
    __syncthreads();
#endif
    for (int e = t; e < 256; e += 128) {
        int rv = __brev((unsigned)e) >> 24;
        float2 z; z.x = s_raw[rv]; z.y = 0.f;
        s[e] = z;
    }
    __syncthreads();
    for (int len = 2; len <= 256; len <<= 1) {
        int half = len >> 1;
        int blk = t / half, pos = t % half;
        int i0 = blk * len + pos, i1 = i0 + half;
        float ang = -6.28318530717958647692f * (float)pos / (float)len;
        float2 w; w.x = __cosf(ang); w.y = __sinf(ang);
        float2 tt = c_mul(w, s[i1]);
        float2 u = s[i0];
        float2 a, bq;
        a.x = u.x + tt.x; a.y = u.y + tt.y;
        bq.x = u.x - tt.x; bq.y = u.y - tt.y;
        s[i0] = a; s[i1] = bq;
        __syncthreads();
    }
    float2* orow = out + ((size_t)b * 256 + y) * 256;
    for (int e = t; e < 256; e += 128) orow[e] = s[e];
}

// Column FFT + |z|^2 + radial-bin accumulation. grid = B*256 blocks, 128 threads.
// Column gather (256 rows x 8B, 2KB stride) done by the Tensor Data Mover.
__global__ void fft_cols_power_kernel(const float2* __restrict__ rows,
                                      float* __restrict__ binsum) {
    int b  = blockIdx.x >> 8;
    int kx = blockIdx.x & 255;
    int t  = threadIdx.x;
    __shared__ float2 s_raw2[256];
    __shared__ float2 s[256];
    const float2* colbase = rows + (size_t)b * 65536 + kx;
#ifdef CDNA5_TDM
    if (threadIdx.x == 0)
        tdm_load_2d(colbase, s_raw2, 3u, 1u, 256u, 1u, 256u, 256u);
    __builtin_amdgcn_s_wait_tensorcnt(0);
    __syncthreads();
#else
    for (int e = t; e < 256; e += 128) s_raw2[e] = colbase[(size_t)e * 256];
    __syncthreads();
#endif
    for (int e = t; e < 256; e += 128) {
        int rv = __brev((unsigned)e) >> 24;
        s[e] = s_raw2[rv];
    }
    __syncthreads();
    for (int len = 2; len <= 256; len <<= 1) {
        int half = len >> 1;
        int blk = t / half, pos = t % half;
        int i0 = blk * len + pos, i1 = i0 + half;
        float ang = -6.28318530717958647692f * (float)pos / (float)len;
        float2 w; w.x = __cosf(ang); w.y = __sinf(ang);
        float2 tt = c_mul(w, s[i1]);
        float2 u = s[i0];
        float2 a, bq;
        a.x = u.x + tt.x; a.y = u.y + tt.y;
        bq.x = u.x - tt.x; bq.y = u.y - tt.y;
        s[i0] = a; s[i1] = bq;
        __syncthreads();
    }
    int fx = (kx < 128) ? kx : kx - 256;
    for (int e = t; e < 256; e += 128) {
        float2 z = s[e];
        float p = z.x * z.x + z.y * z.y;
        int fy = (e < 128) ? e : e - 256;
        float R = sqrtf((float)(fx * fx + fy * fy));
        int bin = (int)(R * 0.5f);        // R / min(cx,cy) * 64 = R/2, floor
        if (bin > 63) bin = 63;
        atomicAdd(&binsum[b * 64 + bin], p);
    }
}

__global__ void ps_counts_kernel(float* __restrict__ counts) {
    int idx = blockIdx.x * 256 + threadIdx.x;   // 65536 positions
    int ky = idx >> 8, kx = idx & 255;
    int fy = (ky < 128) ? ky : ky - 256;
    int fx = (kx < 128) ? kx : kx - 256;
    float R = sqrtf((float)(fx * fx + fy * fy));
    int bin = (int)(R * 0.5f);
    if (bin > 63) bin = 63;
    atomicAdd(&counts[bin], 1.0f);
}

// per-image: ps = log10(sum/count), then (ps-mean)/std. grid = B blocks, 64 threads.
__global__ void ps_norm_kernel(const float* __restrict__ binsum,
                               const float* __restrict__ counts,
                               float* __restrict__ psout) {
    int b = blockIdx.x, t = threadIdx.x;
    __shared__ float red[64];
    float cnt = counts[t];
    float sm  = binsum[b * 64 + t];
    float val = (cnt > 0.f) ? sm / fmaxf(cnt, 1.f) : 0.f;
    float ps = log10f(val + 1e-10f);
    red[t] = ps; __syncthreads();
    for (int o = 32; o > 0; o >>= 1) { if (t < o) red[t] += red[t + o]; __syncthreads(); }
    float mean = red[0] / 64.f;
    __syncthreads();
    float d = ps - mean;
    red[t] = d * d; __syncthreads();
    for (int o = 32; o > 0; o >>= 1) { if (t < o) red[t] += red[t + o]; __syncthreads(); }
    float stdv = sqrtf(red[0] / 63.f) + 1e-10f;
    psout[b * 64 + t] = (ps - mean) / stdv;
}

__global__ void zero_kernel(float* __restrict__ p, int n) {
    int i = blockIdx.x * 256 + threadIdx.x;
    if (i < n) p[i] = 0.f;
}

// ---------------------------------------------------------------------------
// Implicit-GEMM conv, fused BN (+optional residual/ReLU), f16 WMMA, f32 accum.
// Compile-time KSZ/STRIDE/PAD/log2(Wo): all im2col address math is shifts and
// constant-multiply divides. Block tile 64 Cout x 64 pixels, 8 waves (4Mx2N),
// K unrolled by 2 -> 4 v_wmma per barrier.  H(=W input) = Wo*STRIDE.
// LDS staging pre-swizzled to WMMA per-lane operand order (ISA 7.12.2).
// ---------------------------------------------------------------------------
template<int KSZ, int STRIDE, int PAD, int WOL2>
__global__ void conv_wmma_kernel(const float* __restrict__ in,
                                 const float* __restrict__ wgt,
                                 const float* __restrict__ bng, const float* __restrict__ bnb,
                                 const float* __restrict__ bnm, const float* __restrict__ bnv,
                                 const float* __restrict__ res,
                                 float* __restrict__ out,
                                 int Cin, int Cout, int doRelu) {
    constexpr int Wo   = 1 << WOL2;
    constexpr int Npix = Wo * Wo;
    constexpr int H    = Wo * STRIDE;
    constexpr int KK2  = KSZ * KSZ;

    __shared__ __align__(32) _Float16 a_lds[2][4][32][16];
    __shared__ __align__(32) _Float16 b_lds[2][4][32][16];

    const int blockN = blockIdx.x * 64;
    const int cobase = blockIdx.y * 64;
    const int bb     = blockIdx.z;
    const int Ktot   = Cin * KK2;

    const int lane = threadIdx.x & 31;
    const int wave = threadIdx.x >> 5;
    const int wm = wave >> 1;      // 0..3  M subtile
    const int wn = wave & 1;       // 0..1  -> N subtiles {2wn, 2wn+1}

    v8f acc0 = {};
    v8f acc1 = {};

    for (int kc = 0; kc < Ktot; kc += 64) {
        if (kc + 64 < Ktot)        // global_prefetch_b8 next weight K-chunk
            __builtin_prefetch(wgt + (size_t)cobase * Ktot + kc + 64, 0, 1);

        // ---- stage A: weights, WMMA A-operand lane order ----------------
        for (int e = threadIdx.x; e < 4096; e += 256) {
            int kch = e >> 11;
            int sub = (e >> 9) & 3;
            int r   = e & 511;
            int ln  = r >> 4;
            int i   = r & 15;
            int m   = ln & 15;
            int kl  = ((ln >> 4) * 8) + ((i < 8) ? i : (i + 8));
            int kk  = kc + kch * 32 + kl;
            int co  = cobase + sub * 16 + m;
            float v = (kk < Ktot) ? wgt[(size_t)co * Ktot + kk] : 0.f;
            a_lds[kch][sub][ln][i] = (_Float16)v;
        }
        // ---- stage B: implicit im2col, WMMA B-operand lane order --------
        for (int e = threadIdx.x; e < 4096; e += 256) {
            int kch = e >> 11;
            int sub = (e >> 9) & 3;
            int r   = e & 511;
            int ln  = r >> 4;
            int i   = r & 15;
            int nl  = sub * 16 + (ln & 15);
            int kl  = ((ln >> 4) * 8) + ((i < 8) ? i : (i + 8));
            int kk  = kc + kch * 32 + kl;
            float v = 0.f;
            if (kk < Ktot) {
                int ci  = kk / KK2;
                int rem = kk % KK2;
                int ky  = rem / KSZ;
                int kx  = rem % KSZ;
                int p   = blockN + nl;
                int oy  = p >> WOL2, ox = p & (Wo - 1);
                int iy  = oy * STRIDE + ky - PAD;
                int ix  = ox * STRIDE + kx - PAD;
                if ((unsigned)iy < (unsigned)H && (unsigned)ix < (unsigned)H)
                    v = in[(((size_t)bb * Cin + ci) * H + iy) * H + ix];
            }
            b_lds[kch][sub][ln][i] = (_Float16)v;
        }
        __syncthreads();

        v16h a0  = *(const v16h*)(&a_lds[0][wm][lane][0]);
        v16h a1  = *(const v16h*)(&a_lds[1][wm][lane][0]);
        v16h b00 = *(const v16h*)(&b_lds[0][wn * 2 + 0][lane][0]);
        v16h b01 = *(const v16h*)(&b_lds[0][wn * 2 + 1][lane][0]);
        v16h b10 = *(const v16h*)(&b_lds[1][wn * 2 + 0][lane][0]);
        v16h b11 = *(const v16h*)(&b_lds[1][wn * 2 + 1][lane][0]);
        acc0 = __builtin_amdgcn_wmma_f32_16x16x32_f16(false, a0, false, b00, (short)0, acc0, false, false);
        acc1 = __builtin_amdgcn_wmma_f32_16x16x32_f16(false, a0, false, b01, (short)0, acc1, false, false);
        acc0 = __builtin_amdgcn_wmma_f32_16x16x32_f16(false, a1, false, b10, (short)0, acc0, false, false);
        acc1 = __builtin_amdgcn_wmma_f32_16x16x32_f16(false, a1, false, b11, (short)0, acc1, false, false);
        __syncthreads();
    }

    // ---- epilogue: BN fold, residual, ReLU, f32 store ----------------------
    union { v8f v; float f[8]; } u0, u1;
    u0.v = acc0; u1.v = acc1;
    int p0 = blockN + (wn * 2) * 16 + (lane & 15);
    for (int v = 0; v < 8; ++v) {
        int co = cobase + wm * 16 + ((lane >> 4) * 8) + v;
        float sc = bng[co] * rsqrtf(bnv[co] + 1e-5f);
        float sh = bnb[co] - bnm[co] * sc;
        size_t base = ((size_t)bb * Cout + co) * (size_t)Npix;
        float x0 = u0.f[v] * sc + sh;
        float x1 = u1.f[v] * sc + sh;
        if (res) { x0 += res[base + p0]; x1 += res[base + p0 + 16]; }
        if (doRelu) { x0 = fmaxf(x0, 0.f); x1 = fmaxf(x1, 0.f); }
        out[base + p0]      = x0;
        out[base + p0 + 16] = x1;
    }
}

// ---------------------------------------------------------------------------
// Maxpool 3x3 s2 pad1 on [64,64,128,128] -> [64,64,64,64]
// ---------------------------------------------------------------------------
__global__ void maxpool_kernel(const float* __restrict__ in, float* __restrict__ out) {
    size_t i = (size_t)blockIdx.x * 256 + threadIdx.x;
    if (i >= (size_t)64 * 64 * 64 * 64) return;
    int x = i & 63, y = (i >> 6) & 63, c = (i >> 12) & 63, b = (int)(i >> 18);
    float m = -INFINITY;
    for (int dy = 0; dy < 3; ++dy)
        for (int dx = 0; dx < 3; ++dx) {
            int iy = y * 2 + dy - 1, ix = x * 2 + dx - 1;
            if (iy >= 0 && iy < 128 && ix >= 0 && ix < 128)
                m = fmaxf(m, in[(((size_t)b * 64 + c) * 128 + iy) * 128 + ix]);
        }
    out[i] = m;
}

// ---------------------------------------------------------------------------
// Global average pool over H*W: grid = B*C blocks, 256 threads
// ---------------------------------------------------------------------------
__global__ void avgpool_kernel(const float* __restrict__ in, float* __restrict__ out,
                               int C, int HW, int ldo) {
    int bc = blockIdx.x, t = threadIdx.x;
    __shared__ float red[256];
    const float* p = in + (size_t)bc * HW;
    float s = 0.f;
    for (int e = t; e < HW; e += 256) s += p[e];
    red[t] = s; __syncthreads();
    for (int o = 128; o > 0; o >>= 1) { if (t < o) red[t] += red[t + o]; __syncthreads(); }
    if (t == 0) out[(size_t)(bc / C) * ldo + (bc % C)] = red[0] / (float)HW;
}

__global__ void se_scale_kernel(float* __restrict__ x, const float* __restrict__ s,
                                int C, int HW, size_t total) {
    size_t i = (size_t)blockIdx.x * 256 + threadIdx.x;
    if (i >= total) return;
    int c = (int)((i / HW) % C);
    int b = (int)(i / ((size_t)HW * C));
    x[i] *= s[b * C + c];
}

// ---------------------------------------------------------------------------
// Small dense layer: y[b, ycol0+j] = act(x[b,:Din] . W[j,:] + bias[j])
// act: 0=none 1=relu 2=sigmoid
// ---------------------------------------------------------------------------
__global__ void dense_kernel(const float* __restrict__ x, int ldx,
                             const float* __restrict__ Wm, const float* __restrict__ bias,
                             float* __restrict__ y, int ldy, int ycol0,
                             int B, int Din, int Dout, int act) {
    int t = blockIdx.x * 256 + threadIdx.x;
    if (t >= B * Dout) return;
    int b = t / Dout, j = t % Dout;
    const float* xr = x + (size_t)b * ldx;
    const float* wr = Wm + (size_t)j * Din;
    float s = bias ? bias[j] : 0.f;
    for (int i = 0; i < Din; ++i) s += xr[i] * wr[i];
    if (act == 1) s = fmaxf(s, 0.f);
    else if (act == 2) s = 1.f / (1.f + expf(-s));
    y[(size_t)b * ldy + ycol0 + j] = s;
}

// ---------------------------------------------------------------------------
// Host-side orchestration
// ---------------------------------------------------------------------------
static void conv_launch(hipStream_t stream,
                        const float* in, const float* w,
                        const float* bnB, const float* bnG,
                        const float* bnM, const float* bnV,
                        const float* res, float* out,
                        int Cin, int Cout, int ksz, int stride, int wol2, int relu) {
    int Npix = 1 << (2 * wol2);
    dim3 grid(Npix / 64, Cout / 64, 64);
    dim3 blk(256);
#define CW(K, S, P, W) conv_wmma_kernel<K, S, P, W><<<grid, blk, 0, stream>>>( \
        in, w, bnG, bnB, bnM, bnV, res, out, Cin, Cout, relu)
    if (ksz == 7) { CW(7, 2, 3, 7); }
    else if (ksz == 1) {
        if (wol2 == 5) CW(1, 2, 0, 5);
        else if (wol2 == 4) CW(1, 2, 0, 4);
        else CW(1, 2, 0, 3);
    } else if (stride == 2) {
        if (wol2 == 5) CW(3, 2, 1, 5);
        else if (wol2 == 4) CW(3, 2, 1, 4);
        else CW(3, 2, 1, 3);
    } else {
        if (wol2 == 6) CW(3, 1, 1, 6);
        else if (wol2 == 5) CW(3, 1, 1, 5);
        else if (wol2 == 4) CW(3, 1, 1, 4);
        else CW(3, 1, 1, 3);
    }
#undef CW
}

extern "C" void kernel_launch(void* const* d_in, const int* in_sizes, int n_in,
                              void* d_out, int out_size, void* d_ws, size_t ws_size,
                              hipStream_t stream) {
    const float* image = (const float*)d_in[0];
    auto F = [&](int i) { return (const float*)d_in[i]; };

    // --- pytree index map (dict keys sorted, DFS) ---------------------------
    const float* attn_fc1[4] = { F(1), F(3), F(5), F(7) };
    const float* attn_fc2[4] = { F(2), F(4), F(6), F(8) };
    const float *fW0 = F(9),  *fb0 = F(10), *fW1 = F(11), *fb1 = F(12),
                *fW2 = F(13), *fb2 = F(14);
    const float *pW0 = F(15), *pb0 = F(16), *pW1 = F(17), *pb1 = F(18),
                *pW2 = F(19), *pb2 = F(20);
    struct Blk { int bn1, bn2, skbn, skw, w1, w2; };
    Blk blk[4][2] = {
        { {21, 25, -1, -1, 29, 30},   {31, 35, -1, -1, 39, 40} },
        { {41, 45, 49, 53, 54, 55},   {56, 60, -1, -1, 64, 65} },
        { {66, 70, 74, 78, 79, 80},   {81, 85, -1, -1, 89, 90} },
        { {91, 95, 99, 103, 104, 105},{106,110, -1, -1,114,115} },
    };
    const float *stem_b = F(116), *stem_g = F(117), *stem_m = F(118), *stem_v = F(119);
    const float* stem_w = F(120);

    // --- workspace layout (floats) -----------------------------------------
    float* ws = (float*)d_ws;
    float*  D    = ws;                        // 8,388,608 (fft buffer, reused as skip identity)
    float2* fftb = (float2*)ws;
    float*  S    = ws + 8388608;              // 67,108,864 stem out
    float*  A    = ws + 75497472;             // 16,777,216
    float*  Bf   = ws + 92274688;             // 16,777,216
    float*  Cf   = ws + 109051904;            // 16,777,216
    float*  SM   = ws + 125829120;
    float* binsum = SM + 0;        // 4096
    float* counts = SM + 4096;     // 64
    float* psbuf  = SM + 4160;     // 4096
    float* h1     = SM + 8256;     // 8192
    float* h2     = SM + 16448;    // 8192
    float* concat = SM + 24640;    // 36864
    float* sey    = SM + 61504;    // 32768
    float* seh    = SM + 94272;    // 2048
    float* ses    = SM + 96320;    // 32768
    float* f1     = SM + 129088;   // 16384
    float* f2     = SM + 145472;   // 4096
    if (ws_size < (size_t)(125829120 + 149568) * sizeof(float)) return;

    const int Bn = 64;

    // === power-spectrum branch =============================================
    zero_kernel<<<(4160 + 255) / 256, 256, 0, stream>>>(binsum, 4160);
    fft_rows_kernel<<<Bn * 256, 128, 0, stream>>>(image, fftb);
    fft_cols_power_kernel<<<Bn * 256, 128, 0, stream>>>(fftb, binsum);
    ps_counts_kernel<<<256, 256, 0, stream>>>(counts);
    ps_norm_kernel<<<Bn, 64, 0, stream>>>(binsum, counts, psbuf);
    dense_kernel<<<(Bn * 128 + 255) / 256, 256, 0, stream>>>(psbuf, 64, pW0, pb0, h1, 128, 0, Bn, 64, 128, 1);
    dense_kernel<<<(Bn * 128 + 255) / 256, 256, 0, stream>>>(h1, 128, pW1, pb1, h2, 128, 0, Bn, 128, 128, 1);
    dense_kernel<<<(Bn * 64  + 255) / 256, 256, 0, stream>>>(h2, 128, pW2, pb2, concat, 576, 512, Bn, 128, 64, 1);

    // === CNN ===============================================================
    // stem: 1->64, 7x7 s2 p3, BN+ReLU, out 128x128 (wol2=7)
    conv_launch(stream, image, stem_w, stem_b, stem_g, stem_m, stem_v,
                nullptr, S, 1, 64, 7, 2, 7, 1);
    maxpool_kernel<<<(64 * 64 * 64 * 64) / 256, 256, 0, stream>>>(S, A);

    int chans[4]   = {64, 128, 256, 512};
    int strides[4] = {1, 2, 2, 2};
    int cins[4]    = {64, 64, 128, 256};

    for (int st = 0; st < 4; ++st) {
        int Cin = cins[st], Cout = chans[st];
        int sR = strides[st];
        int wol2 = 6 - st;                 // output spatial: 64,32,16,8
        // ---- block 0 ----
        const Blk& b0 = blk[st][0];
        const float* ident = A;
        if (b0.skw >= 0) {                 // conv skip + BN (no relu)
            conv_launch(stream, A, F(b0.skw),
                        F(b0.skbn + 0), F(b0.skbn + 1), F(b0.skbn + 2), F(b0.skbn + 3),
                        nullptr, D, Cin, Cout, 1, sR, wol2, 0);
            ident = D;
        }
        conv_launch(stream, A, F(b0.w1),
                    F(b0.bn1 + 0), F(b0.bn1 + 1), F(b0.bn1 + 2), F(b0.bn1 + 3),
                    nullptr, Bf, Cin, Cout, 3, sR, wol2, 1);
        conv_launch(stream, Bf, F(b0.w2),
                    F(b0.bn2 + 0), F(b0.bn2 + 1), F(b0.bn2 + 2), F(b0.bn2 + 3),
                    ident, Cf, Cout, Cout, 3, 1, wol2, 1);
        // ---- block 1 ----
        const Blk& b1 = blk[st][1];
        conv_launch(stream, Cf, F(b1.w1),
                    F(b1.bn1 + 0), F(b1.bn1 + 1), F(b1.bn1 + 2), F(b1.bn1 + 3),
                    nullptr, Bf, Cout, Cout, 3, 1, wol2, 1);
        conv_launch(stream, Bf, F(b1.w2),
                    F(b1.bn2 + 0), F(b1.bn2 + 1), F(b1.bn2 + 2), F(b1.bn2 + 3),
                    Cf, A, Cout, Cout, 3, 1, wol2, 1);
        // ---- SE attention on A ----
        int HW = (1 << wol2) * (1 << wol2), Cr = Cout / 16;
        avgpool_kernel<<<Bn * Cout, 256, 0, stream>>>(A, sey, Cout, HW, Cout);
        dense_kernel<<<(Bn * Cr   + 255) / 256, 256, 0, stream>>>(sey, Cout, attn_fc1[st], nullptr, seh, Cr, 0, Bn, Cout, Cr, 1);
        dense_kernel<<<(Bn * Cout + 255) / 256, 256, 0, stream>>>(seh, Cr, attn_fc2[st], nullptr, ses, Cout, 0, Bn, Cr, Cout, 2);
        size_t total = (size_t)Bn * Cout * HW;
        se_scale_kernel<<<(unsigned)((total + 255) / 256), 256, 0, stream>>>(A, ses, Cout, HW, total);
    }

    // img_feat = mean over 8x8 into concat cols 0..511
    avgpool_kernel<<<Bn * 512, 256, 0, stream>>>(A, concat, 512, 64, 576);

    // fusion head 576->256->64->1
    dense_kernel<<<(Bn * 256 + 255) / 256, 256, 0, stream>>>(concat, 576, fW0, fb0, f1, 256, 0, Bn, 576, 256, 1);
    dense_kernel<<<(Bn * 64  + 255) / 256, 256, 0, stream>>>(f1, 256, fW1, fb1, f2, 64, 0, Bn, 256, 64, 1);
    dense_kernel<<<(Bn * 1   + 255) / 256, 256, 0, stream>>>(f2, 64, fW2, fb2, (float*)d_out, 1, 0, Bn, 64, 1, 0);
}
